// SelfAttention_19653770347087
// MI455X (gfx1250) — matrix operand order repaired and checked by reading, not compiled, not run on hardware
//
#include <hip/hip_runtime.h>
#include <stdint.h>

#define DIM   192
#define HEADS 4
#define ADIM  96
#define HEADC 24
#define REDIM 48
#define BATCH 4
#define NPIX  65536
#define NTOT  (BATCH * NPIX)      // 262144 pixels
#define SM_SCALE 0.2041241452319315f   // 24^-0.5
#define LN_EPS 1e-6f

typedef __attribute__((ext_vector_type(16))) __bf16 v16bf;
typedef __attribute__((ext_vector_type(8)))  float  v8f;

union AFrag { v16bf v; uint32_t u[8]; };
union Acc   { v8f v; float f[8]; };

__device__ inline uint16_t f2bf(float f) {
  uint32_t u = __builtin_bit_cast(uint32_t, f);
  u += 0x7fffu + ((u >> 16) & 1u);          // round-to-nearest-even
  return (uint16_t)(u >> 16);
}
__device__ inline float bf2f(uint16_t h) {
  uint32_t u = ((uint32_t)h) << 16;
  return __builtin_bit_cast(float, u);
}
// A-matrix (16x32, 16-bit) k index of the pair held in VGPR v for lane-half `half`
__device__ inline int kidxA(int v, int half) {
  return (v < 4 ? 2 * v : 16 + 2 * (v - 4)) + half * 8;
}

// ---- CDNA5 async global->LDS (ASYNCcnt) -----------------------------------
// LDS operand = low 32 bits of the flat shared pointer (ISA 10.2: LDS aperture
// address is addr[31:0]); global address = 64-bit VGPR pair, GV mode.
__device__ inline void async_ld_b128(void* lds, const void* gaddr) {
  uint32_t l = (uint32_t)(uintptr_t)lds;
  asm volatile("global_load_async_to_lds_b128 %0, %1, off"
               :: "v"(l), "v"(gaddr) : "memory");
}
__device__ inline void wait_async(int pending) {
  if (pending) asm volatile("s_wait_asynccnt 0x4" ::: "memory");
  else         asm volatile("s_wait_asynccnt 0x0" ::: "memory");
}

// ---------------- K0: weights -> bf16 ----------------
__global__ __launch_bounds__(256) void k_wcvt(const float* qk_w, const float* vs_w,
                                              const float* proj_w,
                                              uint16_t* W1, uint16_t* WP) {
  int i = blockIdx.x * 256 + threadIdx.x;
  if (i < 384 * 192) W1[i] = f2bf(i < 192 * 192 ? qk_w[i] : vs_w[i - 192 * 192]);
  int j = i - 384 * 192;
  if (j >= 0 && j < 192 * 192) WP[j] = f2bf(proj_w[j]);
}

// ---------------- K1: LayerNorm over channels -> bf16, pixel-major [N][192] ----------------
__global__ __launch_bounds__(256) void k_ln(const float* __restrict__ x0,
                                            const float* __restrict__ ln_w,
                                            const float* __restrict__ ln_b,
                                            uint16_t* __restrict__ XN) {
  int n = blockIdx.x * 256 + threadIdx.x;
  int b = n >> 16, p = n & 65535;
  const float* xp = x0 + (size_t)b * DIM * NPIX + p;
  float s = 0.f, ss = 0.f;
  for (int c = 0; c < DIM; ++c) { float v = xp[(size_t)c * NPIX]; s += v; ss += v * v; }
  float mu  = s * (1.f / DIM);
  float var = ss * (1.f / DIM) - mu * mu;
  float inv = rsqrtf(var + LN_EPS);
  uint32_t* dst = (uint32_t*)(XN + (size_t)n * DIM);
  for (int c = 0; c < DIM; c += 2) {
    float v0 = (xp[(size_t)c * NPIX] - mu) * inv * ln_w[c] + ln_b[c];
    float v1 = (xp[(size_t)(c + 1) * NPIX] - mu) * inv * ln_w[c + 1] + ln_b[c + 1];
    dst[c >> 1] = (uint32_t)f2bf(v0) | ((uint32_t)f2bf(v1) << 16);
  }
}

// ---------------- K2: fused qk+vs GEMM: [384x192] x [192xN], WMMA bf16 ----------------
// Async double-buffered LDS staging: A = W1 rows [m][k], B = XN rows [n][k]
// (XN is pixel-major so the B tile is a contiguous memcpy; k-pairs contiguous
//  in LDS for both operands -> ds_load_b128 fragment fetches).
__global__ __launch_bounds__(128) void k_gemm1(const uint16_t* __restrict__ W1,
                                               const uint16_t* __restrict__ XN,
                                               const float* __restrict__ qk_b,
                                               const float* __restrict__ vs_b,
                                               uint16_t* __restrict__ QK,
                                               uint16_t* __restrict__ VS) {
  __shared__ uint16_t As[2][64 * 32];
  __shared__ uint16_t Bs[2][64 * 32];
  const int tid = threadIdx.x, lane = tid & 31, wv = tid >> 5;
  const int half = lane >> 4, l16 = lane & 15;
  const int m0 = blockIdx.y * 64, n0 = blockIdx.x * 64;
  Acc acc[4];
  const v8f zacc = {0.f, 0.f, 0.f, 0.f, 0.f, 0.f, 0.f, 0.f};
  for (int i = 0; i < 4; ++i) acc[i].v = zacc;
  const int rm = tid >> 1, rkb = (tid & 1) * 16;   // row / 16-elem chunk of the 64x32 tile
  const uint16_t* aRow = W1 + (size_t)(m0 + rm) * DIM + rkb;
  const uint16_t* bRow = XN + (size_t)(n0 + rm) * DIM + rkb;
  uint16_t* aDst0 = &As[0][rm * 32 + rkb];
  uint16_t* aDst1 = &As[1][rm * 32 + rkb];
  uint16_t* bDst0 = &Bs[0][rm * 32 + rkb];
  uint16_t* bDst1 = &Bs[1][rm * 32 + rkb];

  // prologue: stage k-slice 0 into buffer 0 (4 async b128 per wave)
  async_ld_b128(aDst0,     aRow);
  async_ld_b128(aDst0 + 8, aRow + 8);
  async_ld_b128(bDst0,     bRow);
  async_ld_b128(bDst0 + 8, bRow + 8);

  for (int ks = 0; ks < 6; ++ks) {
    const int cur = ks & 1;
    if (ks < 5) {                         // issue next slice into the other buffer
      const uint16_t* an = aRow + (ks + 1) * 32;
      const uint16_t* bn = bRow + (ks + 1) * 32;
      uint16_t* ad = (cur ? aDst0 : aDst1);
      uint16_t* bd = (cur ? bDst0 : bDst1);
      async_ld_b128(ad,     an);
      async_ld_b128(ad + 8, an + 8);
      async_ld_b128(bd,     bn);
      async_ld_b128(bd + 8, bn + 8);
    }
    wait_async(ks < 5 ? 4 : 0);           // current slice landed; next may stay in flight
    __syncthreads();
    AFrag a;
    {
      int m = wv * 16 + l16;
      for (int v = 0; v < 8; ++v)
        a.u[v] = *(const uint32_t*)(&As[cur][m * 32 + kidxA(v, half)]);
    }
    for (int nt = 0; nt < 4; ++nt) {
      AFrag bfr;
      int nn = nt * 16 + l16;
      for (int v = 0; v < 8; ++v)
        bfr.u[v] = *(const uint32_t*)(&Bs[cur][nn * 32 + kidxA(v, half)]);
      acc[nt].v = __builtin_amdgcn_wmma_f32_16x16x32_bf16(false, a.v, false, bfr.v,
                                                          (short)0, acc[nt].v, false, false);
    }
    __syncthreads();                      // all waves done reading before slice ks+2 lands
  }
  for (int nt = 0; nt < 4; ++nt) {
    int n = n0 + nt * 16 + l16;
    for (int r = 0; r < 8; ++r) {
      int grow = m0 + wv * 16 + half * 8 + r;
      float bias = grow < 192 ? qk_b[grow] : vs_b[grow - 192];
      uint16_t o = f2bf(acc[nt].f[r] + bias);
      if (grow < 192) QK[(size_t)grow * NTOT + n] = o;
      else            VS[(size_t)(grow - 192) * NTOT + n] = o;
    }
  }
}

// ---------------- K3: depthwise 3x3 (SAME, zero-pad) ----------------
__global__ __launch_bounds__(256) void k_dw(const uint16_t* __restrict__ VS,
                                            const float* __restrict__ dw,
                                            const float* __restrict__ db,
                                            uint16_t* __restrict__ LOC) {
  int c = blockIdx.y;
  int n = blockIdx.x * 256 + threadIdx.x;
  int b = n >> 16, p = n & 65535, h = p >> 8, w = p & 255;
  const uint16_t* vp = VS + (size_t)c * NTOT + (size_t)b * NPIX;
  float acc = 0.f;
  for (int ky = 0; ky < 3; ++ky) {
    int hh = h + ky - 1; if ((unsigned)hh >= 256u) continue;
    for (int kx = 0; kx < 3; ++kx) {
      int ww = w + kx - 1; if ((unsigned)ww >= 256u) continue;
      acc += bf2f(vp[hh * 256 + ww]) * dw[c * 9 + ky * 3 + kx];
    }
  }
  LOC[(size_t)c * NTOT + n] = f2bf(acc + db[c]);
}

// ---------------- K4: shifted-window attention, one wave32 per (window, head) ----------------
__global__ __launch_bounds__(64) void k_attn(const uint16_t* __restrict__ QK,
                                             const uint16_t* __restrict__ VS,
                                             uint16_t* __restrict__ Y) {
  __shared__ uint16_t Qs[2][64][32];   // [tok][ch], ch padded 24->32
  __shared__ uint16_t Ks[2][64][32];
  __shared__ uint16_t Vt[2][32][64];   // transposed: [ch][tok] -> B k-pairs contiguous
  __shared__ uint16_t Ps[2][64][64];   // softmax(P) bf16
  const int tid = threadIdx.x, lane = tid & 31, wv = tid >> 5;
  const int half = lane >> 4, l16 = lane & 15;
  const int win = blockIdx.x;
  const int b = win >> 10, wy = (win >> 5) & 31, wx = win & 31;
  const int hh = blockIdx.y * 2 + wv;
  const int chq = hh * HEADC, chk = ADIM + hh * HEADC, chv = hh * HEADC;

  // gather window with shift roll(-4,-4) applied: src pixel = (coord+4) mod 256
  for (int hf = 0; hf < 2; ++hf) {
    int t = hf * 32 + lane;
    int ty = t >> 3, tx = t & 7;
    int gh = (wy * 8 + ty + 4) & 255, gw = (wx * 8 + tx + 4) & 255;
    size_t n = (size_t)b * NPIX + gh * 256 + gw;
    for (int c = 0; c < HEADC; ++c) {
      Qs[wv][t][c] = QK[(size_t)(chq + c) * NTOT + n];
      Ks[wv][t][c] = QK[(size_t)(chk + c) * NTOT + n];
      Vt[wv][c][t] = VS[(size_t)(chv + c) * NTOT + n];
    }
    for (int c = HEADC; c < 32; ++c) { Qs[wv][t][c] = 0; Ks[wv][t][c] = 0; Vt[wv][c][t] = 0; }
  }

  const v8f zacc = {0.f, 0.f, 0.f, 0.f, 0.f, 0.f, 0.f, 0.f};
  // S = Q K^T (scaled), row-softmax -> Ps, one 16-row block at a time
  for (int mt = 0; mt < 4; ++mt) {
    AFrag a;
    int m = mt * 16 + l16;
    for (int v = 0; v < 8; ++v) a.u[v] = *(const uint32_t*)&Qs[wv][m][kidxA(v, half)];
    Acc s[4];
    for (int nt = 0; nt < 4; ++nt) {
      AFrag bf;
      int nn = nt * 16 + l16;
      for (int v = 0; v < 8; ++v) bf.u[v] = *(const uint32_t*)&Ks[wv][nn][kidxA(v, half)];
      s[nt].v = __builtin_amdgcn_wmma_f32_16x16x32_bf16(false, a.v, false, bf.v,
                                                        (short)0, zacc, false, false);
    }
    for (int r = 0; r < 8; ++r) {
      int row = mt * 16 + half * 8 + r;
      float sv[4], mx = -1e30f;
      for (int nt = 0; nt < 4; ++nt) { sv[nt] = s[nt].f[r] * SM_SCALE; mx = fmaxf(mx, sv[nt]); }
      for (int off = 1; off < 16; off <<= 1) mx = fmaxf(mx, __shfl_xor(mx, off));
      float pv[4], sum = 0.f;
      for (int nt = 0; nt < 4; ++nt) { pv[nt] = __expf(sv[nt] - mx); sum += pv[nt]; }
      for (int off = 1; off < 16; off <<= 1) sum += __shfl_xor(sum, off);
      float inv = 1.f / sum;
      for (int nt = 0; nt < 4; ++nt) Ps[wv][row][nt * 16 + l16] = f2bf(pv[nt] * inv);
    }
  }

  // O = P V  (K = 64 tokens, two 32-steps; N = 32 padded channels)
  Acc o[4][2];
  for (int mt = 0; mt < 4; ++mt) for (int nt = 0; nt < 2; ++nt) o[mt][nt].v = zacc;
  for (int ks = 0; ks < 2; ++ks) {
    for (int mt = 0; mt < 4; ++mt) {
      AFrag a;
      int m = mt * 16 + l16;
      for (int v = 0; v < 8; ++v) a.u[v] = *(const uint32_t*)&Ps[wv][m][ks * 32 + kidxA(v, half)];
      for (int nt = 0; nt < 2; ++nt) {
        AFrag bf;
        int nn = nt * 16 + l16;
        for (int v = 0; v < 8; ++v)
          bf.u[v] = *(const uint32_t*)&Vt[wv][nn][ks * 32 + kidxA(v, half)];
        o[mt][nt].v = __builtin_amdgcn_wmma_f32_16x16x32_bf16(false, a.v, false, bf.v,
                                                              (short)0, o[mt][nt].v, false, false);
      }
    }
  }

  // scatter back with roll(+4,+4) undone (same src-pixel mapping)
  for (int mt = 0; mt < 4; ++mt) {
    for (int r = 0; r < 8; ++r) {
      int t = mt * 16 + half * 8 + r;
      int ty = t >> 3, tx = t & 7;
      int gh = (wy * 8 + ty + 4) & 255, gw = (wx * 8 + tx + 4) & 255;
      size_t n = (size_t)b * NPIX + gh * 256 + gw;
      for (int nt = 0; nt < 2; ++nt) {
        int c = nt * 16 + l16;
        if (c < HEADC) Y[(size_t)(chq + c) * NTOT + n] = f2bf(o[mt][nt].f[r]);
      }
    }
  }
}

// ---------------- K5: per-(b,c) spatial mean of (y + local) ----------------
__global__ __launch_bounds__(256) void k_stats(const uint16_t* __restrict__ Y,
                                               const uint16_t* __restrict__ VS,
                                               const uint16_t* __restrict__ LOC,
                                               float* __restrict__ S) {
  __shared__ float red[256];
  int c = blockIdx.x, b = blockIdx.y, tid = threadIdx.x;
  const uint16_t* yp = (c < ADIM ? Y : VS) + (size_t)c * NTOT + (size_t)b * NPIX;
  const uint16_t* lp = LOC + (size_t)c * NTOT + (size_t)b * NPIX;
  float acc = 0.f;
  for (int p = tid; p < NPIX; p += 256) acc += bf2f(yp[p]) + bf2f(lp[p]);
  red[tid] = acc; __syncthreads();
  for (int s = 128; s > 0; s >>= 1) { if (tid < s) red[tid] += red[tid + s]; __syncthreads(); }
  if (tid == 0) S[b * DIM + c] = red[0] * (1.f / NPIX);
}

// ---------------- K6: squeeze-excite MLP -> channel attention ----------------
__global__ __launch_bounds__(64) void k_mlp(const float* __restrict__ S,
                                            const float* __restrict__ fu1,
                                            const float* __restrict__ fu2,
                                            float* __restrict__ CA) {
  __shared__ float hsh[REDIM];
  int b = blockIdx.x, tid = threadIdx.x;
  if (tid < REDIM) {
    float acc = 0.f;
    for (int c = 0; c < DIM; ++c) acc += fu1[tid * DIM + c] * S[b * DIM + c];
    hsh[tid] = acc > 0.f ? acc : 0.2f * acc;   // leaky_relu(0.2)
  }
  __syncthreads();
  for (int c = tid; c < DIM; c += 64) {
    float acc = 0.f;
    for (int j = 0; j < REDIM; ++j) acc += fu2[c * REDIM + j] * hsh[j];
    CA[b * DIM + c] = 1.f / (1.f + __expf(-acc));
  }
}

// ---------------- K7: proj GEMM with fused blend epilogue-in / residual-out ----------------
__global__ __launch_bounds__(128) void k_gemm2(const uint16_t* __restrict__ WP,
                                               const uint16_t* __restrict__ Y,
                                               const uint16_t* __restrict__ VS,
                                               const uint16_t* __restrict__ LOC,
                                               const float* __restrict__ CA,
                                               const float* __restrict__ proj_b,
                                               const float* __restrict__ x0,
                                               float* __restrict__ out) {
  __shared__ uint16_t As[64 * 32];
  __shared__ uint16_t Bs[64 * 32];
  const int tid = threadIdx.x, lane = tid & 31, wv = tid >> 5;
  const int half = lane >> 4, l16 = lane & 15;
  const int m0 = blockIdx.y * 64, n0 = blockIdx.x * 64;
  const int bb = n0 >> 16;   // tile never crosses a batch boundary (64 | 65536)
  Acc acc[4];
  const v8f zacc = {0.f, 0.f, 0.f, 0.f, 0.f, 0.f, 0.f, 0.f};
  for (int i = 0; i < 4; ++i) acc[i].v = zacc;
  const int am = tid >> 1, akb = (tid & 1) * 16;
  const int bn = tid & 63, bkb = (tid >> 6) * 16;
  for (int k0 = 0; k0 < DIM; k0 += 32) {
    { // weight tile: async global->LDS, natural [m][k] layout
      const uint16_t* src = WP + (size_t)(m0 + am) * DIM + k0 + akb;
      uint16_t* dst = As + am * 32 + akb;
      async_ld_b128(dst,     src);
      async_ld_b128(dst + 8, src + 8);
    }
    { // B tile: z = y*ca + local*(1-ca), computed on the fly, transposed [n][k]
      uint16_t tmp[16];
      for (int i = 0; i < 16; ++i) {
        int kg = k0 + bkb + i;
        size_t off = (size_t)kg * NTOT + n0 + bn;
        float ca = CA[bb * DIM + kg];
        float yv = bf2f(kg < ADIM ? Y[off] : VS[off]);
        float lv = bf2f(LOC[off]);
        tmp[i] = f2bf(yv * ca + lv * (1.f - ca));
      }
      if (k0 + 32 < DIM)
        __builtin_prefetch(LOC + (size_t)(k0 + 32 + bkb) * NTOT + n0 + bn, 0, 1);
      uint4* dst = (uint4*)(Bs + bn * 32 + bkb);
      dst[0] = ((uint4*)tmp)[0]; dst[1] = ((uint4*)tmp)[1];
    }
    wait_async(0);
    __syncthreads();
    AFrag a;
    {
      int m = wv * 16 + l16;
      for (int v = 0; v < 8; ++v) a.u[v] = *(const uint32_t*)(As + m * 32 + kidxA(v, half));
    }
    for (int nt = 0; nt < 4; ++nt) {
      AFrag bfr;
      int nn = nt * 16 + l16;
      for (int v = 0; v < 8; ++v) bfr.u[v] = *(const uint32_t*)(Bs + nn * 32 + kidxA(v, half));
      acc[nt].v = __builtin_amdgcn_wmma_f32_16x16x32_bf16(false, a.v, false, bfr.v,
                                                          (short)0, acc[nt].v, false, false);
    }
    __syncthreads();
  }
  for (int nt = 0; nt < 4; ++nt) {
    int n = n0 + nt * 16 + l16;
    int p = n & 65535;
    for (int r = 0; r < 8; ++r) {
      int grow = m0 + wv * 16 + half * 8 + r;
      size_t oi = ((size_t)(bb * DIM + grow) << 16) + p;
      out[oi] = acc[nt].f[r] + proj_b[grow] + x0[oi];
    }
  }
}

extern "C" void kernel_launch(void* const* d_in, const int* in_sizes, int n_in,
                              void* d_out, int out_size, void* d_ws, size_t ws_size,
                              hipStream_t stream) {
  (void)in_sizes; (void)n_in; (void)out_size; (void)ws_size;
  const float* x0      = (const float*)d_in[0];
  const float* ln_w    = (const float*)d_in[1];
  const float* ln_b    = (const float*)d_in[2];
  const float* qk_w    = (const float*)d_in[3];
  const float* qk_b    = (const float*)d_in[4];
  const float* vs_w    = (const float*)d_in[5];
  const float* vs_b    = (const float*)d_in[6];
  const float* depth_w = (const float*)d_in[7];
  const float* depth_b = (const float*)d_in[8];
  const float* fu1_w   = (const float*)d_in[9];
  const float* fu2_w   = (const float*)d_in[10];
  const float* proj_w  = (const float*)d_in[11];
  const float* proj_b  = (const float*)d_in[12];
  float* out = (float*)d_out;

  uint16_t* XN  = (uint16_t*)d_ws;                 // [N][192] bf16 pixel-major; reused as Y
  uint16_t* QK  = XN  + (size_t)DIM * NTOT;        // [192][N]: q rows 0..95, k rows 96..191
  uint16_t* VS  = QK  + (size_t)DIM * NTOT;        // [192][N]
  uint16_t* LOC = VS  + (size_t)DIM * NTOT;        // [192][N]
  uint16_t* W1  = LOC + (size_t)DIM * NTOT;        // [384][192] bf16 (qk_w ; vs_w)
  uint16_t* WP  = W1  + 384 * 192;                 // [192][192] bf16
  float*    S   = (float*)(WP + 192 * 192);        // [4][192]
  float*    CA  = S + BATCH * DIM;                 // [4][192]
  uint16_t* Y   = XN;                              // reuse XN (dead after k_gemm1)

  k_wcvt <<<432, 256, 0, stream>>>(qk_w, vs_w, proj_w, W1, WP);
  k_ln   <<<NTOT / 256, 256, 0, stream>>>(x0, ln_w, ln_b, XN);
  k_gemm1<<<dim3(NTOT / 64, 6), 128, 0, stream>>>(W1, XN, qk_b, vs_b, QK, VS);
  k_dw   <<<dim3(NTOT / 256, DIM), 256, 0, stream>>>(VS, depth_w, depth_b, LOC);
  k_attn <<<dim3(BATCH * 32 * 32, 2), 64, 0, stream>>>(QK, VS, Y);
  k_stats<<<dim3(DIM, BATCH), 256, 0, stream>>>(Y, VS, LOC, S);
  k_mlp  <<<BATCH, 64, 0, stream>>>(S, fu1_w, fu2_w, CA);
  k_gemm2<<<dim3(NTOT / 64, 3), 128, 0, stream>>>(WP, Y, VS, LOC, CA, proj_b, x0, out);
}
